// LSTM_76304388981150
// MI455X (gfx1250) — compile-verified, matrix operand order checked by reading, not run on hardware
//
#include <hip/hip_runtime.h>

#define TSTEPS 512
#define NBATCH 32
#define DMODEL 1024
#define HDIM   1024
#define PDIM   1024
#define NLAYER 4
#define G4H    4096
#define RBLOCKS 64

typedef __bf16 bf16_t;
typedef __attribute__((ext_vector_type(16))) __bf16 v16bf;
typedef __attribute__((ext_vector_type(8)))  __bf16 v8bf;
typedef __attribute__((ext_vector_type(8)))  float  v8f;
typedef __attribute__((ext_vector_type(4)))  int    v4i;
typedef __attribute__((ext_vector_type(8)))  int    v8i;
typedef __attribute__((ext_vector_type(4)))  unsigned int v4u;

#if __has_builtin(__builtin_amdgcn_global_load_async_to_lds_b128)
#define HAVE_ASYNC_LDS 1
#endif
#if __has_builtin(__builtin_amdgcn_s_wait_asynccnt)
#define ASYNC_WAIT(n) __builtin_amdgcn_s_wait_asynccnt(n)
#else
#define ASYNC_WAIT(n)
#endif
#if __has_builtin(__builtin_amdgcn_tensor_load_to_lds) && \
    __has_builtin(__builtin_amdgcn_s_wait_tensorcnt)
#define HAVE_TDM 1
#endif

static __device__ __forceinline__ float sigmoidf_(float x) {
  return 1.0f / (1.0f + __expf(-x));
}

// One 16-byte global -> LDS transfer (async if available; ASYNCcnt-tracked).
static __device__ __forceinline__ void copy16_g2l(const bf16_t* __restrict__ g,
                                                  bf16_t* l) {
#ifdef HAVE_ASYNC_LDS
  __builtin_amdgcn_global_load_async_to_lds_b128(
      (__attribute__((address_space(1))) v4i*)(void*)g,
      (__attribute__((address_space(3))) v4i*)(void*)l, 0, 0);
#else
  *reinterpret_cast<v8bf*>(l) = *reinterpret_cast<const v8bf*>(g);
#endif
}

static __device__ __forceinline__ unsigned lds_offset_of(const void* p) {
  return (unsigned)(unsigned long long)(__attribute__((address_space(3))) const char*)p;
}

#ifdef HAVE_TDM
// TDM: DMA a 2D bf16 tile (tile_rows x tile_cols, global row stride = stride
// elements) from global memory into LDS (packed row-major). One instruction
// per issuing wave; completion tracked by TENSORcnt.
// D# layout per CDNA5 ISA ch.8: group0 = control/lds/global addr, group1 =
// dims/strides, remaining groups unused (2D tensor) -> zeros.
// This toolchain's builtin takes 6 args (probe-confirmed):
//   (uint32x4 g0, int32x8 g1, int32x4, int32x4, int32x8, i32 cpol)
static __device__ __forceinline__ void tdm_load_2d(const bf16_t* g, unsigned lds_off,
                                                   int tile_cols, int tile_rows,
                                                   int stride) {
  const unsigned long long ga = (unsigned long long)(const void*)g;
  v4u g0;
  g0.x = 1u;                                        // count=1 (valid), user mode
  g0.y = lds_off;                                   // lds_addr (bytes)
  g0.z = (unsigned)(ga & 0xFFFFFFFFu);              // global_addr[31:0]
  g0.w = (unsigned)((ga >> 32) & 0x01FFFFFFu)       // global_addr[56:32]
         | (2u << 30);                              // type=2 ("image")
  v8i g1;
  g1[0] = (int)(1u << 16);                          // wg_mask=0, data_size=1 (2B)
  g1[1] = (int)((unsigned)(stride & 0xFFFF) << 16); // tensor_dim0[15:0] @ [63:48]
  g1[2] = (int)(((unsigned)stride >> 16) & 0xFFFFu) // tensor_dim0[31:16]
          | (int)((unsigned)(tile_rows & 0xFFFF) << 16); // tensor_dim1[15:0]
  g1[3] = (int)((unsigned)(tile_cols & 0xFFFF) << 16);   // tile_dim0 @ [127:112]
  g1[4] = (int)(unsigned)(tile_rows & 0xFFFF);      // tile_dim1; tile_dim2=0
  g1[5] = stride;                                   // tensor_dim0_stride[31:0]
  g1[6] = 0;                                        // stride hi / dim1_stride lo
  g1[7] = 0;
  const v4i z4 = {0, 0, 0, 0};
  const v8i z8 = {0, 0, 0, 0, 0, 0, 0, 0};
  __builtin_amdgcn_tensor_load_to_lds(g0, g1, z4, z4, z8, 0);
}
#endif

// Load a 16x32 bf16 tile whose 16 "rows" are contiguous along K (stride ld)
// into the wave32 WMMA operand layout for v_wmma_f32_16x16x32_bf16:
//  lanes 0-15  : row = row0+lane,    K = {0..7}  and {16..23}
//  lanes 16-31 : row = row0+lane-16, K = {8..15} and {24..31}
static __device__ __forceinline__ v16bf load_k32_tile(const bf16_t* __restrict__ base,
                                                      int ld, int row0, int k0) {
  const int lane = threadIdx.x & 31;
  const int half = lane >> 4;
  const int r    = row0 + (lane & 15);
  const bf16_t* p = base + (size_t)r * ld + (size_t)k0 + half * 8;
  v8bf lo = *reinterpret_cast<const v8bf*>(p);
  v8bf hi = *reinterpret_cast<const v8bf*>(p + 16);
  v16bf out;
#pragma unroll
  for (int i = 0; i < 8; ++i) { out[i] = lo[i]; out[i + 8] = hi[i]; }
  return out;
}

// ---------------------------------------------------------------------------
// Weight prep: fp32 (K, N) row-major  ->  bf16 (N, K) row-major (LDS-tiled)
// ---------------------------------------------------------------------------
__global__ __launch_bounds__(256) void transpose_cast_kernel(
    const float* __restrict__ src, bf16_t* __restrict__ dst, int rows, int cols) {
  __shared__ float tile[32][33];
  const int bx = blockIdx.x * 32;  // col tile
  const int by = blockIdx.y * 32;  // row tile
  const int tx = threadIdx.x & 31;
  const int ty = threadIdx.x >> 5; // 0..7
#pragma unroll
  for (int i = 0; i < 4; ++i)
    tile[ty + i * 8][tx] = src[(size_t)(by + ty + i * 8) * cols + bx + tx];
  __syncthreads();
#pragma unroll
  for (int i = 0; i < 4; ++i)
    dst[(size_t)(bx + ty + i * 8) * rows + by + tx] = (bf16_t)tile[tx][ty + i * 8];
}

__global__ void bias_add_kernel(const float* __restrict__ a, const float* __restrict__ b,
                                float* __restrict__ o, int n) {
  int i = blockIdx.x * blockDim.x + threadIdx.x;
  if (i < n) o[i] = a[i] + b[i];
}

__global__ void cast_f32_bf16_kernel(const float* __restrict__ src,
                                     bf16_t* __restrict__ dst, int n) {
  for (int i = blockIdx.x * blockDim.x + threadIdx.x; i < n; i += gridDim.x * blockDim.x)
    dst[i] = (bf16_t)src[i];
}

// ---------------------------------------------------------------------------
// gates_x = X @ Wih^T(bf16) + bias   (M=16384, N=4096, K=1024), bf16 out.
// 128 threads = 4 waves; wave w owns a 16(M) x 128(N) strip (8 accumulators).
// The shared 128x32 B chunk is staged once per k-step into double-buffered
// LDS via async global->LDS copies (4x traffic reduction, ASYNCcnt-pipelined).
// ---------------------------------------------------------------------------
#define BCHUNK (128 * 32)

static __device__ __forceinline__ void stage_b_chunk(const bf16_t* __restrict__ BT,
                                                     int n0, int k0, bf16_t* l) {
#pragma unroll
  for (int i = 0; i < 4; ++i) {
    const int t = (int)threadIdx.x + i * 128;  // 0..511
    const int row = t >> 2;                    // 4 transfers per row
    const int co  = (t & 3) * 8;
    copy16_g2l(BT + (size_t)(n0 + row) * 1024 + k0 + co, l + row * 32 + co);
  }
}

__global__ __launch_bounds__(128) void gates_gemm_kernel(
    const bf16_t* __restrict__ A,    // (16384, 1024)
    const bf16_t* __restrict__ BT,   // (4096, 1024)  Wih transposed
    const float*  __restrict__ bias, // (4096)
    bf16_t* __restrict__ out)        // (16384, 4096)
{
  __shared__ __align__(16) bf16_t bstage[2 * BCHUNK];  // 16 KB double buffer
  const int wave = threadIdx.x >> 5;
  const int lane = threadIdx.x & 31;
  const int m0 = blockIdx.x * 64 + wave * 16;
  const int n0 = blockIdx.y * 128;
  bf16_t* sb0 = &bstage[0];        // buffer i = sb0 + (i&1)*BCHUNK (no ptr array:
                                   // LDS-address aggregates break lld static init)
  v8f acc[8] = {};
  stage_b_chunk(BT, n0, 0, sb0);
  for (int kk = 0; kk < 32; ++kk) {
    if (kk < 31) {
      stage_b_chunk(BT, n0, (kk + 1) * 32, sb0 + ((kk + 1) & 1) * BCHUNK);
      ASYNC_WAIT(4);                 // chunk kk complete, kk+1 in flight
    } else {
      ASYNC_WAIT(0);
    }
    __syncthreads();
    v16bf a = load_k32_tile(A, 1024, m0, kk * 32);
    const bf16_t* bb = sb0 + (kk & 1) * BCHUNK;
#pragma unroll
    for (int t = 0; t < 8; ++t) {
      v16bf b = load_k32_tile(bb, 32, t * 16, 0);
      acc[t] = __builtin_amdgcn_wmma_f32_16x16x32_bf16(
          false, a, false, b, (short)0, acc[t], false, false);
    }
    __syncthreads();                 // buffer reusable for chunk kk+2
  }
  const int half = lane >> 4;
  const int colb = lane & 15;
#pragma unroll
  for (int t = 0; t < 8; ++t) {
    const int col = n0 + t * 16 + colb;
    const float bv = bias[col];
#pragma unroll
    for (int r = 0; r < 8; ++r) {
      const int row = m0 + r + half * 8;
      out[(size_t)row * G4H + col] = (bf16_t)(acc[t][r] + bv);
    }
  }
}

// ---------------------------------------------------------------------------
// Persistent recurrent kernel: 64 blocks x 256 threads (8 waves).
// Block b owns hidden columns [16b, 16b+16) for all four gates + projection.
// h is staged into LDS in double-buffered 128-wide K-chunks via the Tensor
// Data Mover (TENSORcnt) -- async-LDS or sync copies as fallback.
// Projection is split-K across all 8 waves.
// ---------------------------------------------------------------------------
static __device__ __forceinline__ void grid_sync(unsigned* cnt, unsigned& expect) {
  __syncthreads();
  expect += RBLOCKS;
  if (threadIdx.x == 0) {
    __hip_atomic_fetch_add(cnt, 1u, __ATOMIC_ACQ_REL, __HIP_MEMORY_SCOPE_AGENT);
    while (__hip_atomic_load(cnt, __ATOMIC_ACQUIRE, __HIP_MEMORY_SCOPE_AGENT) < expect)
      __builtin_amdgcn_s_sleep(1);
  }
  __syncthreads();
}

#define ACHUNK (32 * 128)

// Fallback stage: (32 rows x 128 cols) bf16 chunk (row stride 1024) -> LDS.
static __device__ __forceinline__ void stage_chunk(const bf16_t* __restrict__ g,
                                                   bf16_t* l) {
#pragma unroll
  for (int i = 0; i < 2; ++i) {
    const int t = (int)threadIdx.x + i * 256;  // 0..511
    const int row = t >> 4;
    const int co  = (t & 15) * 8;
    copy16_g2l(g + (size_t)row * 1024 + co, l + row * 128 + co);
  }
}

__global__ __launch_bounds__(256) void recurrent_kernel(
    const bf16_t* __restrict__ gatesx,  // (T*32, 4096) = x@Wih + b
    const bf16_t* __restrict__ whhT,    // (4096, 1024)
    const bf16_t* __restrict__ whmT,    // (1024, 1024)
    const float*  __restrict__ wic,
    const float*  __restrict__ wfc,
    const float*  __restrict__ woc,
    bf16_t* __restrict__ hbuf,          // (32, 1024) current h (bf16)
    bf16_t* __restrict__ mbuf,          // (32, 1024) o*tanh(c) (bf16)
    float*  __restrict__ cstate,        // (32, 1024)
    float*  __restrict__ ys,            // (T, 32, 1024) layer output
    unsigned* __restrict__ cnt)
{
  __shared__ __align__(16) bf16_t astage[2 * ACHUNK]; // 16 KB: h k-chunk double buffer
  __shared__ float pre[4][NBATCH][16];                //  8 KB: i/f/g/o pre-activations
  __shared__ float ppart[8][8][32];                   //  8 KB: split-K proj partials
  const int wave = threadIdx.x >> 5;
  const int lane = threadIdx.x & 31;
  const int half = lane >> 4;
  const int colb = lane & 15;
  const int j0 = blockIdx.x * 16;       // hidden-column slice
  unsigned expect = 0;

  for (int t = 0; t < TSTEPS; ++t) {
    // ---- Phase 1: a_hh slice; each wave one 16x16 tile, K=1024 ----
    {
      const int g_ = wave >> 1;   // gate 0..3
      const int mh = wave & 1;    // batch half
      v8f acc = {};
      const bf16_t* bbase = whhT + (size_t)(g_ * HDIM + j0) * 1024;
      bf16_t* a0 = &astage[0];    // buffer i = a0 + (i&1)*ACHUNK
#ifdef HAVE_TDM
      if (wave == 0) tdm_load_2d(hbuf, lds_offset_of(a0), 128, 32, 1024);
#else
      stage_chunk(hbuf, a0);
#endif
      for (int c = 0; c < 8; ++c) {
#ifdef HAVE_TDM
        if (wave == 0) {
          if (c < 7) {
            tdm_load_2d(hbuf + (c + 1) * 128,
                        lds_offset_of(a0 + ((c + 1) & 1) * ACHUNK), 128, 32, 1024);
            __builtin_amdgcn_s_wait_tensorcnt(1);  // chunk c done, c+1 in flight
          } else {
            __builtin_amdgcn_s_wait_tensorcnt(0);
          }
        }
#else
        if (c < 7) {
          stage_chunk(hbuf + (c + 1) * 128, a0 + ((c + 1) & 1) * ACHUNK);
          ASYNC_WAIT(2);
        } else {
          ASYNC_WAIT(0);
        }
#endif
        __syncthreads();
        const bf16_t* abuf = a0 + (c & 1) * ACHUNK;
#pragma unroll
        for (int kk = 0; kk < 4; ++kk) {
          v16bf a = load_k32_tile(abuf, 128, mh * 16, kk * 32);
          v16bf b = load_k32_tile(bbase, 1024, 0, c * 128 + kk * 32);
          acc = __builtin_amdgcn_wmma_f32_16x16x32_bf16(
              false, a, false, b, (short)0, acc, false, false);
        }
        __syncthreads();                // buf reusable for chunk c+2
      }
      const bf16_t* gx = gatesx + (size_t)t * NBATCH * G4H;
#pragma unroll
      for (int r = 0; r < 8; ++r) {
        const int batch = mh * 16 + r + half * 8;
        pre[g_][batch][colb] =
            acc[r] + (float)gx[(size_t)batch * G4H + g_ * HDIM + j0 + colb];
      }
    }
    __syncthreads();

    // ---- Elementwise LSTM cell for the block's 32x16 slice ----
    for (int idx = threadIdx.x; idx < NBATCH * 16; idx += 256) {
      const int b_ = idx >> 4;
      const int c_ = idx & 15;
      const int j  = j0 + c_;
      float cc = cstate[b_ * HDIM + j];
      const float ai = pre[0][b_][c_], af = pre[1][b_][c_];
      const float ag = pre[2][b_][c_], ao = pre[3][b_][c_];
      const float ig = sigmoidf_(ai + cc * wic[j]);
      const float fg = sigmoidf_(af + cc * wfc[j]);
      const float gg = tanhf(ag);
      const float nc = fg * cc + ig * gg;
      const float og = sigmoidf_(ao + nc * woc[j]);
      cstate[b_ * HDIM + j] = nc;
      mbuf[b_ * HDIM + j]   = (bf16_t)(og * tanhf(nc));
    }
    grid_sync(cnt, expect);  // all of m ready

    // ---- Phase 2: projection, split-K over all 8 waves (K=256 each) ----
    {
      const int mh = wave & 1;
      const int kc = wave >> 1;  // 0..3
      v8f acc = {};
      for (int kk = 0; kk < 8; ++kk) {
        const int k0 = kc * 256 + kk * 32;
        v16bf a = load_k32_tile(mbuf, 1024, mh * 16, k0);
        v16bf b = load_k32_tile(whmT, 1024, j0, k0);
        acc = __builtin_amdgcn_wmma_f32_16x16x32_bf16(
            false, a, false, b, (short)0, acc, false, false);
      }
#pragma unroll
      for (int r = 0; r < 8; ++r) ppart[wave][r][lane] = acc[r];
    }
    __syncthreads();
    // reduce 4 K-partials per output tile, write ys and next-h
    {
      float* yt = ys + (size_t)t * NBATCH * PDIM;
      for (int idx = threadIdx.x; idx < 512; idx += 256) {
        const int mh  = idx >> 8;
        const int rem = idx & 255;
        const int r   = rem >> 5;
        const int ln  = rem & 31;
        const float v = ppart[mh][r][ln] + ppart[2 + mh][r][ln] +
                        ppart[4 + mh][r][ln] + ppart[6 + mh][r][ln];
        const int batch = mh * 16 + r + (ln >> 4) * 8;
        const int col   = j0 + (ln & 15);
        yt[batch * PDIM + col]   = v;
        hbuf[batch * PDIM + col] = (bf16_t)v;
      }
    }
    grid_sync(cnt, expect);  // h ready for next step
  }
}

__global__ void init_state_kernel(const float* __restrict__ h0l, const float* __restrict__ c0l,
                                  bf16_t* __restrict__ hbuf, float* __restrict__ cstate,
                                  unsigned* __restrict__ cnt) {
  if (blockIdx.x == 0 && threadIdx.x == 0) *cnt = 0u;
  for (int i = blockIdx.x * blockDim.x + threadIdx.x; i < NBATCH * PDIM;
       i += gridDim.x * blockDim.x)
    hbuf[i] = (bf16_t)h0l[i];
  for (int i = blockIdx.x * blockDim.x + threadIdx.x; i < NBATCH * HDIM;
       i += gridDim.x * blockDim.x)
    cstate[i] = c0l[i];
}

__global__ void tail_kernel(const float* __restrict__ yslast, const float* __restrict__ cstate,
                            float* __restrict__ outH, float* __restrict__ outC) {
  for (int i = blockIdx.x * blockDim.x + threadIdx.x; i < NBATCH * PDIM;
       i += gridDim.x * blockDim.x) {
    outH[i] = yslast[i];
    outC[i] = cstate[i];
  }
}

// ---------------------------------------------------------------------------
extern "C" void kernel_launch(void* const* d_in, const int* in_sizes, int n_in,
                              void* d_out, int out_size, void* d_ws, size_t ws_size,
                              hipStream_t stream) {
  const float* x    = (const float*)d_in[0];
  const float* h0   = (const float*)d_in[1];
  const float* c0   = (const float*)d_in[2];
  const float* w_ih = (const float*)d_in[3];
  const float* w_hh = (const float*)d_in[4];
  const float* b_ih = (const float*)d_in[5];
  const float* b_hh = (const float*)d_in[6];
  const float* w_ic = (const float*)d_in[7];
  const float* w_fc = (const float*)d_in[8];
  const float* w_oc = (const float*)d_in[9];
  const float* w_hm = (const float*)d_in[10];

  char* p = (char*)d_ws;
  auto carve = [&](size_t bytes) -> char* {
    char* r = p;
    p += (bytes + 255) & ~(size_t)255;
    return r;
  };
  bf16_t* wihT   = (bf16_t*)carve((size_t)NLAYER * G4H * DMODEL * 2);  // 32 MB
  bf16_t* whhT   = (bf16_t*)carve((size_t)NLAYER * G4H * PDIM * 2);    // 32 MB
  bf16_t* whmT   = (bf16_t*)carve((size_t)NLAYER * PDIM * HDIM * 2);   //  8 MB
  float*  bias   = (float*) carve((size_t)NLAYER * G4H * 4);
  bf16_t* xin    = (bf16_t*)carve((size_t)TSTEPS * NBATCH * DMODEL * 2); // 32 MB
  bf16_t* gatesx = (bf16_t*)carve((size_t)TSTEPS * NBATCH * G4H * 2);    // 128 MB
  bf16_t* hbuf   = (bf16_t*)carve((size_t)NBATCH * PDIM * 2);
  bf16_t* mbuf   = (bf16_t*)carve((size_t)NBATCH * HDIM * 2);
  float*  cstate = (float*) carve((size_t)NBATCH * HDIM * 4);
  unsigned* cnt  = (unsigned*)carve(256);

  // One-time (per call) weight prep: fp32 -> transposed bf16 (fits in L2).
  for (int l = 0; l < NLAYER; ++l) {
    transpose_cast_kernel<<<dim3(G4H / 32, DMODEL / 32), 256, 0, stream>>>(
        w_ih + (size_t)l * DMODEL * G4H, wihT + (size_t)l * G4H * DMODEL, DMODEL, G4H);
    transpose_cast_kernel<<<dim3(G4H / 32, PDIM / 32), 256, 0, stream>>>(
        w_hh + (size_t)l * PDIM * G4H, whhT + (size_t)l * G4H * PDIM, PDIM, G4H);
    transpose_cast_kernel<<<dim3(PDIM / 32, HDIM / 32), 256, 0, stream>>>(
        w_hm + (size_t)l * HDIM * PDIM, whmT + (size_t)l * PDIM * HDIM, HDIM, PDIM);
  }
  bias_add_kernel<<<(NLAYER * G4H) / 256, 256, 0, stream>>>(b_ih, b_hh, bias, NLAYER * G4H);

  float* ys   = (float*)d_out;                       // (T, N, P)
  float* outH = ys + (size_t)TSTEPS * NBATCH * PDIM; // (L, N, P)
  float* outC = outH + (size_t)NLAYER * NBATCH * PDIM;

  for (int l = 0; l < NLAYER; ++l) {
    const float* lin = (l == 0) ? x : ys;  // previous layer's ys (stream-ordered reuse)
    cast_f32_bf16_kernel<<<2048, 256, 0, stream>>>(lin, xin, TSTEPS * NBATCH * DMODEL);
    gates_gemm_kernel<<<dim3((TSTEPS * NBATCH) / 64, G4H / 128), 128, 0, stream>>>(
        xin, wihT + (size_t)l * G4H * DMODEL, bias + (size_t)l * G4H, gatesx);
    init_state_kernel<<<64, 256, 0, stream>>>(
        h0 + (size_t)l * NBATCH * PDIM, c0 + (size_t)l * NBATCH * HDIM, hbuf, cstate, cnt);
    recurrent_kernel<<<RBLOCKS, 256, 0, stream>>>(
        gatesx, whhT + (size_t)l * G4H * PDIM, whmT + (size_t)l * PDIM * HDIM,
        w_ic + (size_t)l * HDIM, w_fc + (size_t)l * HDIM, w_oc + (size_t)l * HDIM,
        hbuf, mbuf, cstate, ys, cnt);
    tail_kernel<<<64, 256, 0, stream>>>(
        ys + (size_t)(TSTEPS - 1) * NBATCH * PDIM, cstate,
        outH + (size_t)l * NBATCH * PDIM, outC + (size_t)l * NBATCH * HDIM);
  }
}
